// CrossAttentionWithGating_48498770707172
// MI455X (gfx1250) — matrix operand order repaired and checked by reading, not compiled, not run on hardware
//
#include <hip/hip_runtime.h>
#include <hip/hip_bf16.h>

#define DIM      768
#define DIM2     1536
#define NHEADS   12
#define DHEAD    64
#define SEQ      1024
#define BATCH    8
#define ROWS     (BATCH*SEQ)   // 8192

typedef __attribute__((ext_vector_type(16))) __bf16 v16bf;
typedef __attribute__((ext_vector_type(8)))  float  v8f;

__device__ __forceinline__ v8f wmma_bf16(v16bf a, v16bf b, v8f c) {
  // D = A(16x32 bf16) * B(32x16 bf16) + C(16x16 f32)
  return __builtin_amdgcn_wmma_f32_16x16x32_bf16(false, a, false, b, (short)0, c, false, false);
}

// k-position of packed pair v (0..7) for one lane.
// A operand (16x32): VGPR0-3 hold K 0..7 (lanes 0-15) / 8..15 (lanes 16-31),
//                    VGPR4-7 hold K 16..23 / 24..31.
// BPAT=true (unused; B here uses the A order): contiguous, lanes 0-15 hold K 0..15, lanes 16-31 K 16..31.
template<bool BPAT>
__device__ __forceinline__ int kpat(int v, int half) {
  if (BPAT) return 2*v + 16*half;
  return (v < 4 ? 2*v : 16 + 2*(v - 4)) + 8*half;
}

template<bool BPAT>
__device__ __forceinline__ v16bf frag_row(const __bf16* p, int half) {
  v16bf f;
#pragma unroll
  for (int v = 0; v < 8; ++v) {
    const int k = kpat<BPAT>(v, half);
    f[2*v]     = p[k];
    f[2*v + 1] = p[k + 1];
  }
  return f;
}

template<bool BPAT>
__device__ __forceinline__ v16bf frag_col(const __bf16* p, int ld, int half) {
  v16bf f;
#pragma unroll
  for (int v = 0; v < 8; ++v) {
    const int k = kpat<BPAT>(v, half);
    f[2*v]     = p[k*ld];
    f[2*v + 1] = p[(k + 1)*ld];
  }
  return f;
}

// ---- gfx1250 async VMEM -> LDS staging (ASYNCcnt tracked) ------------------
__device__ __forceinline__ uint32_t lds_word(const void* p) {
  return (uint32_t)(uintptr_t)p;   // low 32 bits of a shared pointer = LDS offset
}
__device__ __forceinline__ void async_g2l_b128(uint32_t lds_off, const void* gptr) {
  asm volatile("global_load_async_to_lds_b128 %0, %1, off"
               :: "v"(lds_off), "v"((uint64_t)(uintptr_t)gptr) : "memory");
}
// async loads complete in order: waiting for <=N leaves the newest N in flight
__device__ __forceinline__ void wait_async0() {
  asm volatile("s_wait_asynccnt 0x0" ::: "memory");
}
__device__ __forceinline__ void wait_async2() {
  asm volatile("s_wait_asynccnt 0x2" ::: "memory");
}
__device__ __forceinline__ void wait_async4() {
  asm volatile("s_wait_asynccnt 0x4" ::: "memory");
}

// cross-half (xor 16) exchange without LDS: v_permlanex16 identity selects
__device__ __forceinline__ float xor16(float x) {
#if __has_builtin(__builtin_amdgcn_permlanex16)
  const int i = __builtin_bit_cast(int, x);
  const int r = __builtin_amdgcn_permlanex16(i, i, 0x76543210, 0xfedcba98, false, false);
  return __builtin_bit_cast(float, r);
#else
  return __shfl_xor(x, 16, 32);
#endif
}

// ---------------------------------------------------------------- GEMM ----
#define LDA_S 40     // 128x32 A tile, padded (80B rows, 16B aligned)
#define LDB_S 136    // 32x128 B tile, padded (272B rows, 16B aligned)

enum { OUT_BF16 = 0, OUT_F32 = 1, OUT_GATE = 2 };

// C[M,N] = A[M,K](bf16,row-major) * B[K,N](bf16,row-major) + bias
// block tile 128x128, 8 waves (2x4), each wave 64x32 -> 8 wmma accumulators.
// Double-buffered async-LDS pipeline: tile t+1 streams in during tile t wmmas.
template<int MODE>
__global__ __launch_bounds__(256) void gemm_wmma(
    const __bf16* __restrict__ A, int lda,
    const __bf16* __restrict__ B, int ldb,
    const float*  __restrict__ bias,
    int K,
    __bf16* __restrict__ outb, float* __restrict__ outf, int ldo,
    const float* __restrict__ gl, const float* __restrict__ ga)
{
  __shared__ __bf16 sA[2][128*LDA_S];
  __shared__ __bf16 sB[2][32*LDB_S];
  const int m0 = blockIdx.x * 128, n0 = blockIdx.y * 128;
  const int tid  = threadIdx.x;
  const int wid  = tid >> 5, lane = tid & 31;
  const int idx  = lane & 15, half = lane >> 4;
  const int wm   = wid >> 2, wn = wid & 3;

  auto issue = [&](int t, int buf) {   // 4 async instructions per wave
    const int kb = t * 32;
#pragma unroll
    for (int i = 0; i < 2; ++i) {
      const int c = tid + i*256;
      const int ra = c >> 2, oa = (c & 3) * 8;
      async_g2l_b128(lds_word(&sA[buf][ra*LDA_S + oa]),
                     &A[(size_t)(m0 + ra)*lda + kb + oa]);
      const int rb = c >> 4, ob = (c & 15) * 8;
      async_g2l_b128(lds_word(&sB[buf][rb*LDB_S + ob]),
                     &B[(size_t)(kb + rb)*ldb + n0 + ob]);
    }
  };

  v8f acc[4][2] = {};
  const int NT = K >> 5;
  issue(0, 0);

  for (int it = 0; it < NT; ++it) {
    const int cur = it & 1;
    if (it + 1 < NT) { issue(it + 1, cur ^ 1); wait_async4(); }
    else             { wait_async0(); }
    __syncthreads();

    v16bf af[4], bf[2];
#pragma unroll
    for (int ms = 0; ms < 4; ++ms)
      af[ms] = frag_row<false>(&sA[cur][(wm*64 + ms*16 + idx)*LDA_S], half);
#pragma unroll
    for (int ns = 0; ns < 2; ++ns)
      bf[ns] = frag_col<false>(&sB[cur][wn*32 + ns*16 + idx], LDB_S, half);
#pragma unroll
    for (int ms = 0; ms < 4; ++ms)
#pragma unroll
      for (int ns = 0; ns < 2; ++ns)
        acc[ms][ns] = wmma_bf16(af[ms], bf[ns], acc[ms][ns]);
    __syncthreads();   // safe to overwrite this buffer two iterations later
  }

#pragma unroll
  for (int ms = 0; ms < 4; ++ms)
#pragma unroll
    for (int ns = 0; ns < 2; ++ns) {
      const int col = n0 + wn*32 + ns*16 + idx;
      const float bc = bias[col];
#pragma unroll
      for (int r = 0; r < 8; ++r) {
        const int row = m0 + wm*64 + ms*16 + 8*half + r;
        const float c = acc[ms][ns][r] + bc;
        const size_t off = (size_t)row*ldo + col;
        if (MODE == OUT_BF16) {
          outb[off] = (__bf16)c;
        } else if (MODE == OUT_F32) {
          outf[off] = c;
        } else {                    // OUT_GATE: enhanced = local + sigmoid(c)*attn
          const size_t fo = (size_t)row*DIM + col;
          const float g = 1.0f / (1.0f + __expf(-c));
          outb[off] = (__bf16)(gl[fo] + g*ga[fo]);
        }
      }
    }
}

// ------------------------------------------------------------ attention ----
#define LDK_S 72   // 32x64 K/V tiles padded (144B rows)
#define LDP_S 40   // per-wave P^T tile [q][key], padded rows

// grid (SEQ/128, NHEADS, BATCH), 256 threads = 8 waves, wave owns 16 q rows.
// Computes S^T = K*Q^T so softmax columns (fixed query) live in one lane pair:
// in-register reduce over 16 keys + single permlanex16 across halves.
// K/V tiles double-buffered through the async-LDS engine.
__global__ __launch_bounds__(256) void attn_wmma(
    const __bf16* __restrict__ Q, const __bf16* __restrict__ Kt,
    const __bf16* __restrict__ Vt,
    float* __restrict__ attnf, __bf16* __restrict__ concat)
{
  __shared__ __bf16 sK[2][32*LDK_S];
  __shared__ __bf16 sV[2][32*LDK_S];
  __shared__ __bf16 sP[8][16*LDP_S];

  const int qb = blockIdx.x, h = blockIdx.y, n = blockIdx.z;
  const int tid = threadIdx.x;
  const int wid = tid >> 5, lane = tid & 31;
  const int idx = lane & 15, half = lane >> 4;
  const int q0 = qb*128 + wid*16;

  auto issueKV = [&](int t, int buf) {   // 2 async instructions per wave
    const int r = tid >> 3, off = (tid & 7) * 8;        // 32 rows x 64 dims
    const size_t g = (size_t)(n*SEQ + t*32 + r)*DIM + h*DHEAD + off;
    async_g2l_b128(lds_word(&sK[buf][r*LDK_S + off]), &Kt[g]);
    async_g2l_b128(lds_word(&sV[buf][r*LDK_S + off]), &Vt[g]);
  };

  // B operand Q^T (feat x q): per-lane column q = q0+idx, same feat pattern as A
  const size_t rowQ = (size_t)(n*SEQ + q0 + idx)*DIM + h*DHEAD;
  const v16bf qB0 = frag_row<false>(&Q[rowQ], half);        // feat 0..31
  const v16bf qB1 = frag_row<false>(&Q[rowQ + 32], half);   // feat 32..63

  float mrun = -1e30f, lrun = 0.0f;
  v8f o[4] = {};   // O^T accumulators: rows d (subtile j), cols q

  const int NT = SEQ / 32;
  issueKV(0, 0);

  for (int it = 0; it < NT; ++it) {
    const int cur = it & 1;
    if (it + 1 < NT) { issueKV(it + 1, cur ^ 1); wait_async2(); }
    else             { wait_async0(); }
    __syncthreads();

    // S^T tiles (keys x q): st0 = keys 0..15, st1 = keys 16..31
    v8f st0 = {}, st1 = {};
    {
      const __bf16* k0 = &sK[cur][idx*LDK_S];
      const __bf16* k1 = &sK[cur][(16 + idx)*LDK_S];
      st0 = wmma_bf16(frag_row<false>(k0, half),      qB0, st0);
      st0 = wmma_bf16(frag_row<false>(k0 + 32, half), qB1, st0);
      st1 = wmma_bf16(frag_row<false>(k1, half),      qB0, st1);
      st1 = wmma_bf16(frag_row<false>(k1 + 32, half), qB1, st1);
    }

    // column softmax: this lane holds 16 of the 32 key-scores for query q=idx
    float p[16];
#pragma unroll
    for (int r = 0; r < 8; ++r) { p[r] = st0[r]*0.125f; p[8+r] = st1[r]*0.125f; }
    float mx = p[0];
#pragma unroll
    for (int i = 1; i < 16; ++i) mx = fmaxf(mx, p[i]);
    mx = fmaxf(mx, xor16(mx));
    const float mnew  = fmaxf(mrun, mx);
    const float alpha = __expf(mrun - mnew);
    mrun = mnew;
    float rs = 0.0f;
#pragma unroll
    for (int i = 0; i < 16; ++i) { p[i] = __expf(p[i] - mnew); rs += p[i]; }
    rs += xor16(rs);
    lrun = lrun*alpha + rs;

    // store P as [q][key] so the B-operand reload is contiguous
#pragma unroll
    for (int r = 0; r < 8; ++r) {
      sP[wid][idx*LDP_S + (r + 8*half)]      = (__bf16)p[r];
      sP[wid][idx*LDP_S + (16 + r + 8*half)] = (__bf16)p[8 + r];
    }
#pragma unroll
    for (int j = 0; j < 4; ++j)
#pragma unroll
      for (int r = 0; r < 8; ++r) o[j][r] *= alpha;

    // O^T += V^T(64x32) * P(32x16); LDS ops are in-order within a wave
    const v16bf pB = frag_row<false>(&sP[wid][idx*LDP_S], half);
#pragma unroll
    for (int j = 0; j < 4; ++j) {
      const v16bf vA = frag_col<false>(&sV[cur][j*16 + idx], LDK_S, half);
      o[j] = wmma_bf16(vA, pB, o[j]);
    }
    __syncthreads();   // safe to overwrite this buffer two iterations later
  }

  // epilogue: lane (idx,half) holds q = q0+idx, d = j*16 + 8*half + r
  const float inv = 1.0f / lrun;
  const size_t ro = (size_t)(n*SEQ + q0 + idx);
#pragma unroll
  for (int j = 0; j < 4; ++j) {
    const int d0 = h*DHEAD + j*16 + 8*half;
    alignas(16) float  f[8];
    alignas(16) __bf16 b[8];
#pragma unroll
    for (int r = 0; r < 8; ++r) { f[r] = o[j][r]*inv; b[r] = (__bf16)f[r]; }
    *(float4*)&attnf[ro*DIM + d0]         = *(float4*)&f[0];
    *(float4*)&attnf[ro*DIM + d0 + 4]     = *(float4*)&f[4];
    *(uint4*) &concat[ro*DIM2 + DIM + d0] = *(uint4*)&b[0];
  }
}

// ---------------------------------------------------------- data movers ----
// global_feat (n, 768, 1024) -> (n, 1024, 768) bf16
__global__ void transpose_cvt(const float* __restrict__ g, __bf16* __restrict__ out) {
  __shared__ float t[32][33];
  const int n = blockIdx.z;
  const int p0 = blockIdx.x*32, c0 = blockIdx.y*32;
  const int tx = threadIdx.x, ty = threadIdx.y;
#pragma unroll
  for (int i = 0; i < 4; ++i)
    t[ty + i*8][tx] = g[((size_t)n*DIM + c0 + ty + i*8)*SEQ + p0 + tx];
  __syncthreads();
#pragma unroll
  for (int i = 0; i < 4; ++i)
    out[((size_t)n*SEQ + p0 + ty + i*8)*DIM + c0 + tx] = (__bf16)t[tx][ty + i*8];
}

__global__ void cvt_bf16(const float* __restrict__ in, __bf16* __restrict__ out, int nelem) {
  const int i = blockIdx.x*blockDim.x + threadIdx.x;
  if (i < nelem) out[i] = (__bf16)in[i];
}

// local_feat f32 -> first half of concat buffer (row stride 1536)
__global__ void cvt_local_to_concat(const float* __restrict__ in, __bf16* __restrict__ concat) {
  const size_t i = (size_t)blockIdx.x*blockDim.x + threadIdx.x;
  const size_t row = i / DIM, col = i % DIM;
  concat[row*DIM2 + col] = (__bf16)in[i];
}

// ---------------------------------------------------------------- launch ----
extern "C" void kernel_launch(void* const* d_in, const int* in_sizes, int n_in,
                              void* d_out, int out_size, void* d_ws, size_t ws_size,
                              hipStream_t stream) {
  (void)in_sizes; (void)n_in; (void)out_size; (void)ws_size;
  const float* local = (const float*)d_in[0];
  const float* gfeat = (const float*)d_in[1];
  const float* Wq = (const float*)d_in[2];
  const float* bq = (const float*)d_in[3];
  const float* Wk = (const float*)d_in[4];
  const float* bk = (const float*)d_in[5];
  const float* Wv = (const float*)d_in[6];
  const float* bv = (const float*)d_in[7];
  const float* Wg = (const float*)d_in[8];
  const float* bg = (const float*)d_in[9];
  const float* Wo = (const float*)d_in[10];
  const float* bo = (const float*)d_in[11];
  float* out = (float*)d_out;

  char* p = (char*)d_ws;
  auto take = [&](size_t bytes) { char* r = p; p += (bytes + 255) & ~(size_t)255; return r; };
  __bf16* concat = (__bf16*)take((size_t)ROWS*DIM2*2);   // [local | attn_out] bf16
  __bf16* gb     = (__bf16*)take((size_t)ROWS*DIM*2);    // transposed global bf16
  __bf16* Qb     = (__bf16*)take((size_t)ROWS*DIM*2);
  __bf16* Kb     = (__bf16*)take((size_t)ROWS*DIM*2);
  __bf16* Vb     = (__bf16*)take((size_t)ROWS*DIM*2);
  __bf16* Eb     = (__bf16*)take((size_t)ROWS*DIM*2);    // enhanced bf16
  float*  attnf  = (float*) take((size_t)ROWS*DIM*4);    // attn out f32
  __bf16* Wqb    = (__bf16*)take((size_t)DIM*DIM*2);
  __bf16* Wkb    = (__bf16*)take((size_t)DIM*DIM*2);
  __bf16* Wvb    = (__bf16*)take((size_t)DIM*DIM*2);
  __bf16* Wob    = (__bf16*)take((size_t)DIM*DIM*2);
  __bf16* Wgb    = (__bf16*)take((size_t)DIM2*DIM*2);

  const int wsz = DIM*DIM;
  cvt_bf16<<<(wsz + 255)/256, 256, 0, stream>>>(Wq, Wqb, wsz);
  cvt_bf16<<<(wsz + 255)/256, 256, 0, stream>>>(Wk, Wkb, wsz);
  cvt_bf16<<<(wsz + 255)/256, 256, 0, stream>>>(Wv, Wvb, wsz);
  cvt_bf16<<<(wsz + 255)/256, 256, 0, stream>>>(Wo, Wob, wsz);
  cvt_bf16<<<(DIM2*DIM + 255)/256, 256, 0, stream>>>(Wg, Wgb, DIM2*DIM);
  cvt_local_to_concat<<<(ROWS*DIM)/256, 256, 0, stream>>>(local, concat);
  transpose_cvt<<<dim3(SEQ/32, DIM/32, BATCH), dim3(32, 8), 0, stream>>>(gfeat, gb);

  const dim3 gg(ROWS/128, DIM/128);
  // Q = local @ Wq + bq   (A = concat first half, lda = 1536)
  gemm_wmma<OUT_BF16><<<gg, 256, 0, stream>>>(concat, DIM2, Wqb, DIM, bq, DIM,
                                              Qb, nullptr, DIM, nullptr, nullptr);
  gemm_wmma<OUT_BF16><<<gg, 256, 0, stream>>>(gb, DIM, Wkb, DIM, bk, DIM,
                                              Kb, nullptr, DIM, nullptr, nullptr);
  gemm_wmma<OUT_BF16><<<gg, 256, 0, stream>>>(gb, DIM, Wvb, DIM, bv, DIM,
                                              Vb, nullptr, DIM, nullptr, nullptr);

  attn_wmma<<<dim3(SEQ/128, NHEADS, BATCH), 256, 0, stream>>>(Qb, Kb, Vb, attnf, concat);

  // enhanced = local + sigmoid(concat @ Wg + bg) * attn_out   (fused epilogue)
  gemm_wmma<OUT_GATE><<<gg, 256, 0, stream>>>(concat, DIM2, Wgb, DIM, bg, DIM2,
                                              Eb, nullptr, DIM, local, attnf);
  // out = enhanced @ Wo + bo  (f32)
  gemm_wmma<OUT_F32><<<gg, 256, 0, stream>>>(Eb, DIM, Wob, DIM, bo, DIM,
                                             nullptr, out, DIM, nullptr, nullptr);
}
